// SlotAttention_86998857547867
// MI455X (gfx1250) — matrix operand order repaired and checked by reading, not compiled, not run on hardware
//
#include <hip/hip_runtime.h>
#include <hip/hip_bf16.h>

typedef __attribute__((ext_vector_type(16))) _Float16 v16h;
typedef __attribute__((ext_vector_type(8)))  _Float16 v8h;
typedef __attribute__((ext_vector_type(8)))  float    v8f;

#define B_   32
#define S_   11
#define F_   4096
#define INP_ 512
#define SLOT_ 512
#define KVQ_ 512
#define HID_ 2048
#define SCALE_ 0.044194173824159216f  /* 512^-0.5 */
#define EPS_ 1e-8f
#define LN_EPS_ 1e-5f
#define KSPLIT_U 16   /* split-K for updates GEMM (K=4096) */
#define KSPLIT_M 4    /* split-K for mlp2 GEMM (K=2048) */

#define WMMA_F16(a,b,c) __builtin_amdgcn_wmma_f32_16x16x32_f16(false,(a),false,(b),(short)0,(c),false,false)

__device__ inline v8f zero_v8f() {
  v8f z;
#pragma unroll
  for (int i = 0; i < 8; ++i) z[i] = 0.0f;
  return z;
}

// A fragment, f16 16x32 tile at A (row-major, leading dim lda), k-offset kb.
// ISA layout: lanes 0-15 -> M=lane, K = kb+{0..7, 16..23}; lanes 16-31 -> K+8.
__device__ inline v16h load_a16(const _Float16* __restrict__ A, int lda, int kb) {
  int L = threadIdx.x & 31;
  const _Float16* p = A + (size_t)(L & 15) * lda + kb + ((L >> 4) << 3);
  v8h lo = *(const v8h*)p;
  v8h hi = *(const v8h*)(p + 16);
  v16h a;
#pragma unroll
  for (int i = 0; i < 8; ++i) { a[i] = lo[i]; a[i + 8] = hi[i]; }
  return a;
}

// B fragment from "weight layout" W[n][k] (row n contiguous in k), col block n0, k-offset kb.
// ISA layout: lanes 0-15 -> N=lane, K = kb+{0..7, 16..23}; lanes 16-31 -> K+8.
__device__ inline v16h load_b16(const _Float16* __restrict__ W, int ldk, int n0, int kb) {
  int L = threadIdx.x & 31;
  const _Float16* p = W + (size_t)(n0 + (L & 15)) * ldk + kb + ((L >> 4) << 3);
  v8h lo = *(const v8h*)p;
  v8h hi = *(const v8h*)(p + 16);
  v16h b;
#pragma unroll
  for (int i = 0; i < 8; ++i) { b[i] = lo[i]; b[i + 8] = hi[i]; }
  return b;
}

__device__ inline float wave_sum(float v) {
#pragma unroll
  for (int off = 16; off > 0; off >>= 1) v += __shfl_xor(v, off, 32);
  return v;
}

__device__ inline float sigmoidf_(float x) { return 1.0f / (1.0f + __expf(-x)); }

// ---------------------------------------------------------------- converts
__global__ void k_f32_to_f16(const float* __restrict__ src, _Float16* __restrict__ dst, int n) {
  int i = blockIdx.x * blockDim.x + threadIdx.x;
  if (i < n) dst[i] = (_Float16)src[i];
}

__global__ void k_copy_f32(const float* __restrict__ src, float* __restrict__ dst, int n) {
  int i = blockIdx.x * blockDim.x + threadIdx.x;
  if (i < n) dst[i] = src[i];
}

// ------------------------------------------------- LN(features) -> f16 rows
__global__ void k_ln_feats(const float* __restrict__ x, const float* __restrict__ g,
                           const float* __restrict__ bt, _Float16* __restrict__ out) {
  int row = (blockIdx.x * blockDim.x + threadIdx.x) >> 5;   // B*F rows
  int L = threadIdx.x & 31;
  const float* p = x + (size_t)row * INP_;
  float v[16];
  float s = 0.0f;
#pragma unroll
  for (int i = 0; i < 16; ++i) { v[i] = p[L * 16 + i]; s += v[i]; }
  float mean = wave_sum(s) * (1.0f / INP_);
  float s2 = 0.0f;
#pragma unroll
  for (int i = 0; i < 16; ++i) { float d = v[i] - mean; s2 += d * d; }
  float rstd = __frsqrt_rn(wave_sum(s2) * (1.0f / INP_) + LN_EPS_);
  _Float16* o = out + (size_t)row * INP_;
#pragma unroll
  for (int i = 0; i < 16; ++i) {
    int c = L * 16 + i;
    o[c] = (_Float16)((v[i] - mean) * rstd * g[c] + bt[c]);
  }
}

// ------------------- LN(slot rows) -> f16 padded [B][16][512] (+opt f32)
__global__ void k_ln_slots(const float* __restrict__ slots, const float* __restrict__ g,
                           const float* __restrict__ bt, _Float16* __restrict__ out16,
                           float* __restrict__ outf) {
  int row = (blockIdx.x * blockDim.x + threadIdx.x) >> 5;   // B*16 rows
  int b = row >> 4, s = row & 15;
  int L = threadIdx.x & 31;
  _Float16* o16 = out16 + (size_t)row * SLOT_;
  if (s >= S_) {                                            // zero padding rows
#pragma unroll
    for (int i = 0; i < 16; ++i) o16[L * 16 + i] = (_Float16)0.0f;
    return;
  }
  const float* p = slots + ((size_t)b * S_ + s) * SLOT_;
  float v[16];
  float sm = 0.0f;
#pragma unroll
  for (int i = 0; i < 16; ++i) { v[i] = p[L * 16 + i]; sm += v[i]; }
  float mean = wave_sum(sm) * (1.0f / SLOT_);
  float s2 = 0.0f;
#pragma unroll
  for (int i = 0; i < 16; ++i) { float d = v[i] - mean; s2 += d * d; }
  float rstd = __frsqrt_rn(wave_sum(s2) * (1.0f / SLOT_) + LN_EPS_);
  float* of = outf ? outf + ((size_t)b * S_ + s) * SLOT_ : nullptr;
#pragma unroll
  for (int i = 0; i < 16; ++i) {
    int c = L * 16 + i;
    float y = (v[i] - mean) * rstd * g[c] + bt[c];
    o16[c] = (_Float16)y;
    if (of) of[c] = y;
  }
}

// ------------- keys & valuesT GEMM: [B*F,512] x 2x[512,512] (shared A-frag)
__global__ __launch_bounds__(256)
void k_kv_gemm(const _Float16* __restrict__ feats, const _Float16* __restrict__ wk,
               const _Float16* __restrict__ wv, _Float16* __restrict__ keys,
               _Float16* __restrict__ valsT) {
  int wave = threadIdx.x >> 5;
  int mBase = blockIdx.x * 128 + wave * 16;   // grid.x = (B*F)/128
  int nBase = blockIdx.y * 64;                // grid.y = 512/64
  const _Float16* A = feats + (size_t)mBase * INP_;
  v8f ck[4], cv[4];
#pragma unroll
  for (int t = 0; t < 4; ++t) { ck[t] = zero_v8f(); cv[t] = zero_v8f(); }
  for (int kb = 0; kb < INP_; kb += 32) {
    // near-scope speculative prefetch of next A chunk (no guard: spills into
    // adjacent ws buffer which is always mapped; speculative => safe)
    __builtin_prefetch(A + (size_t)(threadIdx.x & 15) * INP_ + kb + 64, 0, 3);
    v16h a = load_a16(A, INP_, kb);
#pragma unroll
    for (int t = 0; t < 4; ++t) {
      v16h bk = load_b16(wk, INP_, nBase + t * 16, kb);
      ck[t] = WMMA_F16(a, bk, ck[t]);
      v16h bv = load_b16(wv, INP_, nBase + t * 16, kb);
      cv[t] = WMMA_F16(a, bv, cv[t]);
    }
  }
  int L = threadIdx.x & 31;
  int mOff = (L >> 4) << 3;
  int b = mBase >> 12;              // 4096 rows per batch; blocks never straddle
  int fRow = mBase & (F_ - 1);
#pragma unroll
  for (int t = 0; t < 4; ++t) {
    int n = nBase + t * 16 + (L & 15);
#pragma unroll
    for (int r = 0; r < 8; ++r)
      keys[((size_t)mBase + mOff + r) * KVQ_ + n] = (_Float16)ck[t][r];
    _Float16* vp = valsT + ((size_t)b * KVQ_ + n) * F_ + fRow + mOff;
#pragma unroll
    for (int r = 0; r < 8; ++r) vp[r] = (_Float16)cv[t][r];
  }
}

// ----------------------------- queries = sn16 @ Wq^T, f16 padded [B][16][512]
__global__ __launch_bounds__(256)
void k_queries(const _Float16* __restrict__ sn16, const _Float16* __restrict__ wq,
               _Float16* __restrict__ q16) {
  int b = blockIdx.x;
  int nBase = (threadIdx.x >> 5) * 64;        // 8 waves cover 512 cols
  const _Float16* A = sn16 + (size_t)b * 16 * SLOT_;
  v8f acc[4];
#pragma unroll
  for (int t = 0; t < 4; ++t) acc[t] = zero_v8f();
  for (int kb = 0; kb < SLOT_; kb += 32) {
    v16h a = load_a16(A, SLOT_, kb);
#pragma unroll
    for (int t = 0; t < 4; ++t)
      acc[t] = WMMA_F16(a, load_b16(wq, SLOT_, nBase + t * 16, kb), acc[t]);
  }
  int L = threadIdx.x & 31;
  int mOff = (L >> 4) << 3;
  _Float16* Q = q16 + (size_t)b * 16 * KVQ_;
#pragma unroll
  for (int t = 0; t < 4; ++t) {
    int n = nBase + t * 16 + (L & 15);
#pragma unroll
    for (int r = 0; r < 8; ++r) Q[(size_t)(mOff + r) * KVQ_ + n] = (_Float16)acc[t][r];
  }
}

// -------- dots = Q K^T * scale, softmax over slots (in-register), outputs
__global__ __launch_bounds__(256)
void k_dots(const _Float16* __restrict__ q16, const _Float16* __restrict__ keys,
            _Float16* __restrict__ attn16, float* __restrict__ out_attn) {
  int b = blockIdx.x >> 3;                                   // grid.x = B*8
  int fBase = ((blockIdx.x & 7) * 8 + (threadIdx.x >> 5)) * 64;
  const _Float16* A = q16 + (size_t)b * 16 * KVQ_;
  const _Float16* K = keys + (size_t)b * F_ * KVQ_;
  v8f acc[4];
#pragma unroll
  for (int t = 0; t < 4; ++t) acc[t] = zero_v8f();
  for (int kb = 0; kb < KVQ_; kb += 32) {
    v16h a = load_a16(A, KVQ_, kb);
#pragma unroll
    for (int t = 0; t < 4; ++t)
      acc[t] = WMMA_F16(a, load_b16(K, KVQ_, fBase + t * 16, kb), acc[t]);
  }
  int L = threadIdx.x & 31;
  int mOff = (L >> 4) << 3;                                  // lane half: slots 0-7 / 8-15
  _Float16* AT = attn16 + (size_t)b * 16 * F_;
#pragma unroll
  for (int t = 0; t < 4; ++t) {
    int f = fBase + t * 16 + (L & 15);
    float v[8], mx = -1e30f;
#pragma unroll
    for (int r = 0; r < 8; ++r) {
      v[r] = (mOff + r < S_) ? acc[t][r] * SCALE_ : -1e30f;
      mx = fmaxf(mx, v[r]);
    }
    mx = fmaxf(mx, __shfl_xor(mx, 16, 32));                  // reduce over both slot halves
    float e[8], sum = 0.0f;
#pragma unroll
    for (int r = 0; r < 8; ++r) {
      e[r] = (mOff + r < S_) ? __expf(v[r] - mx) : 0.0f;
      sum += e[r];
    }
    sum += __shfl_xor(sum, 16, 32);
    float inv = 1.0f / sum;
#pragma unroll
    for (int r = 0; r < 8; ++r) {
      int m = mOff + r;
      float p = e[r] * inv;                                  // 0 on padded slots
      AT[(size_t)m * F_ + f] = (_Float16)p;
      if (m < S_) out_attn[((size_t)b * S_ + m) * F_ + f] = p;
    }
  }
}

// ----------------------------- feature-axis sums of pre_norm_attn (+ F*eps)
__global__ void k_rowsum(const float* __restrict__ out_attn, float* __restrict__ rowsum) {
  int row = (blockIdx.x * blockDim.x + threadIdx.x) >> 5;    // B*16 rows
  int b = row >> 4, s = row & 15;
  int L = threadIdx.x & 31;
  if (s >= S_) { if (L == 0) rowsum[row] = 1.0f; return; }
  const float* p = out_attn + ((size_t)b * S_ + s) * F_;
  float sum = 0.0f;
  for (int i = L; i < F_; i += 32) sum += p[i];
  sum = wave_sum(sum);
  if (L == 0) rowsum[row] = sum + (float)F_ * EPS_;
}

// ------------ updates split-K partials: attn[16,4096] x V[4096,512] (16-way)
__global__ __launch_bounds__(256)
void k_updates_sk(const _Float16* __restrict__ attn16, const _Float16* __restrict__ valsT,
                  float* __restrict__ updp) {
  int b = blockIdx.x;
  int ks = blockIdx.y;                                       // 0..KSPLIT_U-1
  int nBase = (threadIdx.x >> 5) * 64;
  const _Float16* A = attn16 + (size_t)b * 16 * F_;
  const _Float16* V = valsT + (size_t)b * KVQ_ * F_;
  int k0 = ks * (F_ / KSPLIT_U);
  v8f acc[4];
#pragma unroll
  for (int t = 0; t < 4; ++t) acc[t] = zero_v8f();
  for (int kb = k0; kb < k0 + F_ / KSPLIT_U; kb += 32) {
    v16h a = load_a16(A, F_, kb);
#pragma unroll
    for (int t = 0; t < 4; ++t)
      acc[t] = WMMA_F16(a, load_b16(V, F_, nBase + t * 16, kb), acc[t]);
  }
  int L = threadIdx.x & 31;
  int mOff = (L >> 4) << 3;
  float* P = updp + ((size_t)ks * B_ + b) * 16 * KVQ_;
#pragma unroll
  for (int t = 0; t < 4; ++t) {
    int n = nBase + t * 16 + (L & 15);
#pragma unroll
    for (int r = 0; r < 8; ++r) P[(size_t)(mOff + r) * KVQ_ + n] = acc[t][r];
  }
}

// reduce partials, divide by rowsum, -> f16 padded [B][16][512]
__global__ void k_upd_fin(const float* __restrict__ updp, const float* __restrict__ rowsum,
                          _Float16* __restrict__ upd16) {
  int i = blockIdx.x * blockDim.x + threadIdx.x;             // B*16*512
  int row = i >> 9;
  float s = 0.0f;
#pragma unroll
  for (int ks = 0; ks < KSPLIT_U; ++ks) s += updp[(size_t)ks * (B_ * 16 * KVQ_) + i];
  upd16[i] = (_Float16)(s / rowsum[row]);
}

// --------------------------------- GRU cell: 6 WMMA accumulators per n-tile
__global__ __launch_bounds__(256)
void k_gru(const _Float16* __restrict__ upd16, const _Float16* __restrict__ sn16,
           const float* __restrict__ snf, const _Float16* __restrict__ wih,
           const _Float16* __restrict__ whh, const float* __restrict__ bih,
           const float* __restrict__ bhh, float* __restrict__ slots) {
  int b = blockIdx.x >> 2;                                   // grid.x = B*4
  int n0 = ((blockIdx.x & 3) * 8 + (threadIdx.x >> 5)) * 16; // 32 n-tiles per b
  const _Float16* Au = upd16 + (size_t)b * 16 * KVQ_;
  const _Float16* Ah = sn16 + (size_t)b * 16 * SLOT_;
  v8f axr = zero_v8f(), axz = zero_v8f(), axn = zero_v8f();
  v8f ahr = zero_v8f(), ahz = zero_v8f(), ahn = zero_v8f();
  for (int kb = 0; kb < KVQ_; kb += 32) {
    v16h au = load_a16(Au, KVQ_, kb);
    v16h ah = load_a16(Ah, SLOT_, kb);
    axr = WMMA_F16(au, load_b16(wih, KVQ_, n0, kb), axr);
    axz = WMMA_F16(au, load_b16(wih, KVQ_, n0 + SLOT_, kb), axz);
    axn = WMMA_F16(au, load_b16(wih, KVQ_, n0 + 2 * SLOT_, kb), axn);
    ahr = WMMA_F16(ah, load_b16(whh, SLOT_, n0, kb), ahr);
    ahz = WMMA_F16(ah, load_b16(whh, SLOT_, n0 + SLOT_, kb), ahz);
    ahn = WMMA_F16(ah, load_b16(whh, SLOT_, n0 + 2 * SLOT_, kb), ahn);
  }
  int L = threadIdx.x & 31;
  int mOff = (L >> 4) << 3;
  int n = n0 + (L & 15);
  float bxr = bih[n], bxz = bih[n + SLOT_], bxn = bih[n + 2 * SLOT_];
  float bhr = bhh[n], bhz = bhh[n + SLOT_], bhn = bhh[n + 2 * SLOT_];
#pragma unroll
  for (int r = 0; r < 8; ++r) {
    int m = mOff + r;
    if (m < S_) {
      size_t idx = ((size_t)b * S_ + m) * SLOT_ + n;
      float h = snf[idx];
      float rg = sigmoidf_((axr[r] + bxr) + (ahr[r] + bhr));
      float zg = sigmoidf_((axz[r] + bxz) + (ahz[r] + bhz));
      float ng = tanhf((axn[r] + bxn) + rg * (ahn[r] + bhn));
      slots[idx] = (1.0f - zg) * ng + zg * h;
    }
  }
}

// ------------------------------------- MLP1: [16,512]x[512,2048], ReLU
__global__ __launch_bounds__(256)
void k_mlp1(const _Float16* __restrict__ mln16, const _Float16* __restrict__ w1,
            const float* __restrict__ b1, _Float16* __restrict__ h1) {
  int b = blockIdx.x >> 2;                                   // grid.x = B*4
  int nBase = ((blockIdx.x & 3) * 8 + (threadIdx.x >> 5)) * 64;
  const _Float16* A = mln16 + (size_t)b * 16 * SLOT_;
  v8f acc[4];
#pragma unroll
  for (int t = 0; t < 4; ++t) acc[t] = zero_v8f();
  for (int kb = 0; kb < SLOT_; kb += 32) {
    v16h a = load_a16(A, SLOT_, kb);
#pragma unroll
    for (int t = 0; t < 4; ++t)
      acc[t] = WMMA_F16(a, load_b16(w1, SLOT_, nBase + t * 16, kb), acc[t]);
  }
  int L = threadIdx.x & 31;
  int mOff = (L >> 4) << 3;
  _Float16* H = h1 + (size_t)b * 16 * HID_;
#pragma unroll
  for (int t = 0; t < 4; ++t) {
    int n = nBase + t * 16 + (L & 15);
    float bb = b1[n];
#pragma unroll
    for (int r = 0; r < 8; ++r)
      H[(size_t)(mOff + r) * HID_ + n] = (_Float16)fmaxf(acc[t][r] + bb, 0.0f);
  }
}

// ------------ MLP2 split-K partials: [16,2048]x[2048,512] (4-way)
__global__ __launch_bounds__(256)
void k_mlp2_sk(const _Float16* __restrict__ h1, const _Float16* __restrict__ w2,
               float* __restrict__ h2p) {
  int b = blockIdx.x;
  int ks = blockIdx.y;                                       // 0..KSPLIT_M-1
  int nBase = (threadIdx.x >> 5) * 64;
  const _Float16* A = h1 + (size_t)b * 16 * HID_;
  int k0 = ks * (HID_ / KSPLIT_M);
  v8f acc[4];
#pragma unroll
  for (int t = 0; t < 4; ++t) acc[t] = zero_v8f();
  for (int kb = k0; kb < k0 + HID_ / KSPLIT_M; kb += 32) {
    v16h a = load_a16(A, HID_, kb);
#pragma unroll
    for (int t = 0; t < 4; ++t)
      acc[t] = WMMA_F16(a, load_b16(w2, HID_, nBase + t * 16, kb), acc[t]);
  }
  int L = threadIdx.x & 31;
  int mOff = (L >> 4) << 3;
  float* P = h2p + ((size_t)ks * B_ + b) * 16 * SLOT_;
#pragma unroll
  for (int t = 0; t < 4; ++t) {
    int n = nBase + t * 16 + (L & 15);
#pragma unroll
    for (int r = 0; r < 8; ++r) P[(size_t)(mOff + r) * SLOT_ + n] = acc[t][r];
  }
}

// reduce partials + bias + residual -> slots (and final d_out)
__global__ void k_mlp2_fin(const float* __restrict__ h2p, const float* __restrict__ b2,
                           float* __restrict__ slots, float* __restrict__ out_slots,
                           int write_out) {
  int i = blockIdx.x * blockDim.x + threadIdx.x;             // B*16*512
  int n = i & 511;
  int m = (i >> 9) & 15;
  int b = i >> 13;
  if (m >= S_) return;
  float s = 0.0f;
#pragma unroll
  for (int ks = 0; ks < KSPLIT_M; ++ks) s += h2p[(size_t)ks * (B_ * 16 * SLOT_) + i];
  size_t idx = ((size_t)b * S_ + m) * SLOT_ + n;
  float v = slots[idx] + s + b2[n];
  slots[idx] = v;
  if (write_out) out_slots[idx] = v;
}

// =========================================================================
extern "C" void kernel_launch(void* const* d_in, const int* in_sizes, int n_in,
                              void* d_out, int out_size, void* d_ws, size_t ws_size,
                              hipStream_t stream) {
  const float* in_slots  = (const float*)d_in[0];
  const float* features  = (const float*)d_in[1];
  const float* ln_f_g    = (const float*)d_in[2];
  const float* ln_f_b    = (const float*)d_in[3];
  const float* ln_s_g    = (const float*)d_in[4];
  const float* ln_s_b    = (const float*)d_in[5];
  const float* W_k       = (const float*)d_in[6];
  const float* W_v       = (const float*)d_in[7];
  const float* W_q       = (const float*)d_in[8];
  const float* gru_w_ih  = (const float*)d_in[9];
  const float* gru_w_hh  = (const float*)d_in[10];
  const float* gru_b_ih  = (const float*)d_in[11];
  const float* gru_b_hh  = (const float*)d_in[12];
  const float* ln_m_g    = (const float*)d_in[13];
  const float* ln_m_b    = (const float*)d_in[14];
  const float* mlp_w1    = (const float*)d_in[15];
  const float* mlp_b1    = (const float*)d_in[16];
  const float* mlp_w2    = (const float*)d_in[17];
  const float* mlp_b2    = (const float*)d_in[18];

  float* out_slots = (float*)d_out;                            // [B,11,512]
  float* out_attn  = (float*)d_out + (size_t)B_ * S_ * SLOT_;  // [B,11,4096]

  char* ws = (char*)d_ws;
  size_t off = 0;
  auto alloc = [&](size_t bytes) -> void* {
    void* p = ws + off;
    off += (bytes + 255) & ~(size_t)255;
    return p;
  };
  _Float16* feats16 = (_Float16*)alloc((size_t)B_ * F_ * INP_ * 2);
  _Float16* keys16  = (_Float16*)alloc((size_t)B_ * F_ * KVQ_ * 2);
  _Float16* valsT16 = (_Float16*)alloc((size_t)B_ * KVQ_ * F_ * 2);
  _Float16* wk16    = (_Float16*)alloc((size_t)KVQ_ * INP_ * 2);
  _Float16* wv16    = (_Float16*)alloc((size_t)KVQ_ * INP_ * 2);
  _Float16* wq16    = (_Float16*)alloc((size_t)KVQ_ * SLOT_ * 2);
  _Float16* wih16   = (_Float16*)alloc((size_t)3 * SLOT_ * KVQ_ * 2);
  _Float16* whh16   = (_Float16*)alloc((size_t)3 * SLOT_ * SLOT_ * 2);
  _Float16* w1_16   = (_Float16*)alloc((size_t)HID_ * SLOT_ * 2);
  _Float16* w2_16   = (_Float16*)alloc((size_t)SLOT_ * HID_ * 2);
  float*    slots   = (float*)alloc((size_t)B_ * S_ * SLOT_ * 4);
  float*    snf     = (float*)alloc((size_t)B_ * S_ * SLOT_ * 4);
  _Float16* sn16    = (_Float16*)alloc((size_t)B_ * 16 * SLOT_ * 2);
  _Float16* q16     = (_Float16*)alloc((size_t)B_ * 16 * KVQ_ * 2);
  _Float16* attn16  = (_Float16*)alloc((size_t)B_ * 16 * F_ * 2);
  float*    rowsum  = (float*)alloc((size_t)B_ * 16 * 4);
  _Float16* upd16   = (_Float16*)alloc((size_t)B_ * 16 * KVQ_ * 2);
  _Float16* mln16   = (_Float16*)alloc((size_t)B_ * 16 * SLOT_ * 2);
  _Float16* h1_16   = (_Float16*)alloc((size_t)B_ * 16 * HID_ * 2);
  float*    updp    = (float*)alloc((size_t)KSPLIT_U * B_ * 16 * KVQ_ * 4);
  float*    h2p     = (float*)alloc((size_t)KSPLIT_M * B_ * 16 * SLOT_ * 4);

  auto cvt = [&](const float* s, _Float16* d, int n) {
    k_f32_to_f16<<<(n + 255) / 256, 256, 0, stream>>>(s, d, n);
  };
  cvt(W_k, wk16, KVQ_ * INP_);
  cvt(W_v, wv16, KVQ_ * INP_);
  cvt(W_q, wq16, KVQ_ * SLOT_);
  cvt(gru_w_ih, wih16, 3 * SLOT_ * KVQ_);
  cvt(gru_w_hh, whh16, 3 * SLOT_ * SLOT_);
  cvt(mlp_w1, w1_16, HID_ * SLOT_);
  cvt(mlp_w2, w2_16, SLOT_ * HID_);

  // LN(features) -> f16 ; one wave per row
  k_ln_feats<<<(B_ * F_) / 8, 256, 0, stream>>>(features, ln_f_g, ln_f_b, feats16);

  // keys + valuesT GEMM
  k_kv_gemm<<<dim3((B_ * F_) / 128, KVQ_ / 64), 256, 0, stream>>>(
      feats16, wk16, wv16, keys16, valsT16);

  // working copy of slots
  {
    int n = B_ * S_ * SLOT_;
    k_copy_f32<<<(n + 255) / 256, 256, 0, stream>>>(in_slots, slots, n);
  }

  for (int it = 0; it < 3; ++it) {
    k_ln_slots<<<(B_ * 16) / 8, 256, 0, stream>>>(slots, ln_s_g, ln_s_b, sn16, snf);
    k_queries<<<B_, 256, 0, stream>>>(sn16, wq16, q16);
    k_dots<<<B_ * 8, 256, 0, stream>>>(q16, keys16, attn16, out_attn);
    k_rowsum<<<(B_ * 16) / 8, 256, 0, stream>>>(out_attn, rowsum);
    k_updates_sk<<<dim3(B_, KSPLIT_U), 256, 0, stream>>>(attn16, valsT16, updp);
    k_upd_fin<<<(B_ * 16 * KVQ_) / 256, 256, 0, stream>>>(updp, rowsum, upd16);
    k_gru<<<B_ * 4, 256, 0, stream>>>(upd16, sn16, snf, wih16, whh16,
                                      gru_b_ih, gru_b_hh, slots);
    k_ln_slots<<<(B_ * 16) / 8, 256, 0, stream>>>(slots, ln_m_g, ln_m_b, mln16, nullptr);
    k_mlp1<<<B_ * 4, 256, 0, stream>>>(mln16, w1_16, mlp_b1, h1_16);
    k_mlp2_sk<<<dim3(B_, KSPLIT_M), 256, 0, stream>>>(h1_16, w2_16, h2p);
    k_mlp2_fin<<<(B_ * 16 * SLOT_) / 256, 256, 0, stream>>>(h2p, mlp_b2, slots, out_slots,
                                                            (it == 2) ? 1 : 0);
  }
}
